// DiffuMamba3_89610197663821
// MI455X (gfx1250) — matrix operand order repaired and checked by reading, not compiled, not run on hardware
//
#include <hip/hip_runtime.h>
#include <hip/hip_bf16.h>
#include <math.h>

// Model dims
#define B_      2
#define L_      512
#define V_      50304
#define D_      768
#define NL_     2
#define DI_     1536
#define N_      16
#define DTR_    48
#define H_      1536
#define COND_   128
#define FREQ_   256
#define MASK_ID_ 50257
#define BL_     (B_*L_)
#define NEGF    (-3.4028234663852886e38f)

typedef __attribute__((ext_vector_type(16))) __bf16 v16bf;
typedef __attribute__((ext_vector_type(8)))  float v8f;

// bf16 conversion: round-to-nearest (ties up), 1 VALU op per float;
// pair merge via single v_perm_b32.
__device__ __forceinline__ unsigned pack2(float lo, float hi) {
  union { float f; unsigned u; } a, b; a.f = lo; b.f = hi;
  return __builtin_amdgcn_perm(b.u + 0x8000u, a.u + 0x8000u, 0x07060302u);
}

// Fragment-layout LDS offsets (bf16 element units).
// A subtile 16(M)x32(K): lane = m + 16*(k>>4); elem = k&15
__device__ __forceinline__ int a_off(int sub, int mr, int k) {
  int lane = mr + ((k >> 4) << 4);
  int e = k & 15;
  return sub * 512 + lane * 16 + e;
}
// B frag 32(K)x16(N): lane = n + 16*(k>>4); elem = k&15
__device__ __forceinline__ int b_off(int frag, int nc, int k) {
  int lane = nc + ((k >> 4) << 4);
  int e = k & 15;
  return frag * 512 + lane * 16 + e;
}

// ---- staging helpers (loads fill zero-padded register arrays; all LDS
//      stores are packed 2xbf16 b32 stores, shared by fast & slow paths) ----

// A: 16 k-contiguous floats per thread (row ar, k-half ah)
__device__ __forceinline__ void load_a_regs(const float* __restrict__ A, int M, int K,
                                            int lda, int m0, int k0, int ar, int ah,
                                            bool fullK, float fs[16])
{
  const int gm = m0 + ar;
  if (gm < M && fullK) {
    const float* ap = A + (long)gm * lda + k0 + ah * 16;
    float4 f0 = *(const float4*)(ap);
    float4 f1 = *(const float4*)(ap + 4);
    float4 f2 = *(const float4*)(ap + 8);
    float4 f3 = *(const float4*)(ap + 12);
    fs[0]=f0.x; fs[1]=f0.y; fs[2]=f0.z;  fs[3]=f0.w;
    fs[4]=f1.x; fs[5]=f1.y; fs[6]=f1.z;  fs[7]=f1.w;
    fs[8]=f2.x; fs[9]=f2.y; fs[10]=f2.z; fs[11]=f2.w;
    fs[12]=f3.x;fs[13]=f3.y;fs[14]=f3.z; fs[15]=f3.w;
  } else {
    #pragma unroll
    for (int q = 0; q < 16; ++q) {
      int gk = k0 + ah * 16 + q;
      fs[q] = (gm < M && gk < K) ? A[(long)gm * lda + gk] : 0.0f;
    }
  }
}
__device__ __forceinline__ void store_a_lds(__bf16* lsA, int asub, int amr, int ah,
                                            const float fs[16])
{
  #pragma unroll
  for (int q = 0; q < 8; ++q) {
    int k = ah * 16 + 2 * q;
    *(unsigned*)&lsA[a_off(asub, amr, k)] = pack2(fs[2 * q], fs[2 * q + 1]);
  }
}

// B (row-major K x N): 2 k-rows x 8 n-cols per thread (kp = k-pair, n8 = col base)
__device__ __forceinline__ void load_b_regs(const float* __restrict__ Bm, int N, int K,
                                            int ldb, int n0, int k0, int kp, int n8,
                                            bool fullK, float fs[16])
{
  const int gk = k0 + 2 * kp;
  if (fullK && (n0 + 128 <= N)) {
    const float* p0 = Bm + (long)gk * ldb + n0 + n8;
    const float* p1 = p0 + ldb;
    float4 a0 = *(const float4*)p0,       a1 = *(const float4*)(p0 + 4);
    float4 b0 = *(const float4*)p1,       b1 = *(const float4*)(p1 + 4);
    fs[0]=a0.x; fs[1]=a0.y; fs[2]=a0.z;  fs[3]=a0.w;
    fs[4]=a1.x; fs[5]=a1.y; fs[6]=a1.z;  fs[7]=a1.w;
    fs[8]=b0.x; fs[9]=b0.y; fs[10]=b0.z; fs[11]=b0.w;
    fs[12]=b1.x;fs[13]=b1.y;fs[14]=b1.z; fs[15]=b1.w;
  } else {
    #pragma unroll
    for (int kk = 0; kk < 2; ++kk) {
      #pragma unroll
      for (int c = 0; c < 8; ++c) {
        int gkk = gk + kk, gn = n0 + n8 + c;
        fs[kk * 8 + c] = (gkk < K && gn < N) ? Bm[(long)gkk * ldb + gn] : 0.0f;
      }
    }
  }
}
__device__ __forceinline__ void store_b_lds(__bf16* lsB, int kp, int n8, const float fs[16])
{
  #pragma unroll
  for (int c = 0; c < 8; ++c) {
    int nc = n8 + c;
    *(unsigned*)&lsB[b_off(nc >> 4, nc & 15, 2 * kp)] = pack2(fs[c], fs[8 + c]);
  }
}

// Bt (N x K): 16 k-contiguous floats for one n column (nc, k-half kh16)
__device__ __forceinline__ void load_bt_regs(const float* __restrict__ Bm, int N, int K,
                                             int ldb, int n0, int k0, int nc, int kh16,
                                             bool fullK, float fs[16])
{
  const int gn = n0 + nc;
  if (gn < N && fullK) {
    const float* bp = Bm + (long)gn * ldb + k0 + kh16 * 16;
    float4 f0 = *(const float4*)(bp);
    float4 f1 = *(const float4*)(bp + 4);
    float4 f2 = *(const float4*)(bp + 8);
    float4 f3 = *(const float4*)(bp + 12);
    fs[0]=f0.x; fs[1]=f0.y; fs[2]=f0.z;  fs[3]=f0.w;
    fs[4]=f1.x; fs[5]=f1.y; fs[6]=f1.z;  fs[7]=f1.w;
    fs[8]=f2.x; fs[9]=f2.y; fs[10]=f2.z; fs[11]=f2.w;
    fs[12]=f3.x;fs[13]=f3.y;fs[14]=f3.z; fs[15]=f3.w;
  } else {
    #pragma unroll
    for (int q = 0; q < 16; ++q) {
      int gk = k0 + kh16 * 16 + q;
      fs[q] = (gn < N && gk < K) ? Bm[(long)gn * ldb + gk] : 0.0f;
    }
  }
}
__device__ __forceinline__ void store_bt_lds(__bf16* lsB, int nc, int kh16, const float fs[16])
{
  // base: frag = nc>>4; lane = (nc&15) + 16*kh16; e runs 0..15 (= k&15)
  __bf16* p = lsB + ((nc >> 4) * 512 + ((nc & 15) + (kh16 << 4)) * 16);
  #pragma unroll
  for (int q = 0; q < 8; ++q)
    *(unsigned*)&p[2 * q] = pack2(fs[2 * q], fs[2 * q + 1]);
}

// ---------------------------------------------------------------------------
// GEMM: C[M,N] = A[M,K] @ B[K,N]    (BT=true: B supplied as Bt[N,K])
// f32 in/out, bf16 WMMA f32-accumulate. 256 threads = 8 waves (wave32),
// arranged 4(M) x 2(N). Block tile 128(M) x 128(N), K-step 32.
// Each wave computes a 32x64 strip: 2 A frags x 4 B frags -> 8 v_wmma per
// K-step. Double-buffered LDS pipeline, one barrier per K-step. LDS tiles
// are pre-permuted into WMMA fragment layout: each lane's fragment is 32
// contiguous bytes (2x ds_load_b128).
// ---------------------------------------------------------------------------
template<bool BT>
__global__ __launch_bounds__(256) void gemm_wmma_kernel(
    const float* __restrict__ A, const float* __restrict__ Bm, float* __restrict__ C,
    int M, int N, int K, int lda, int ldb, int ldc)
{
  constexpr int TM = 128, TN = 128, TK = 32;
  __shared__ alignas(32) __bf16 lsA[2][TM * TK];   // 2 x 8 KB
  __shared__ alignas(32) __bf16 lsB[2][TK * TN];   // 2 x 8 KB

  const int tid  = threadIdx.x;
  const int lane = tid & 31;
  const int w    = tid >> 5;
  const int wm   = w >> 1;            // wave M index 0..3 (32 rows each)
  const int wn   = w & 1;             // wave N index 0..1 (64 cols each)
  const int m0   = blockIdx.y * TM;
  const int n0   = blockIdx.x * TN;
  const int mi   = lane & 15;
  const int kh   = lane >> 4;

  // staging assignments
  const int ar   = tid >> 1;          // A row 0..127
  const int ah   = tid & 1;           // A k-half
  const int asub = ar >> 4, amr = ar & 15;
  const int bkp  = tid >> 4;          // B: k-pair 0..15        (non-BT)
  const int bn8  = (tid & 15) * 8;    // B: col base 0..120     (non-BT)
  const int btn  = tid >> 1;          // Bt: n column 0..127    (BT)
  const int bth  = tid & 1;           // Bt: k-half             (BT)

  v8f acc[2][4];
  #pragma unroll
  for (int i = 0; i < 2; ++i)
    #pragma unroll
    for (int j = 0; j < 4; ++j) acc[i][j] = (v8f){};

  const int ksteps = (K + TK - 1) / TK;
  float fa[16];
  float fb[16];

  // prologue: stage step 0 into buffer 0
  {
    bool fullK0 = (TK <= K);
    load_a_regs(A, M, K, lda, m0, 0, ar, ah, fullK0, fa);
    if (!BT) load_b_regs(Bm, N, K, ldb, n0, 0, bkp, bn8, fullK0, fb);
    else     load_bt_regs(Bm, N, K, ldb, n0, 0, btn, bth, fullK0, fb);
    store_a_lds(lsA[0], asub, amr, ah, fa);
    if (!BT) store_b_lds(lsB[0], bkp, bn8, fb);
    else     store_bt_lds(lsB[0], btn, bth, fb);
  }
  __syncthreads();

  for (int s = 0; s < ksteps; ++s) {
    const int cur = s & 1, nxt = cur ^ 1;
    const bool more = (s + 1 < ksteps);

    // prefetch next K-tile from global into registers (overlaps the WMMAs)
    if (more) {
      int k0 = (s + 1) * TK;
      bool fullK = (k0 + TK <= K);
      load_a_regs(A, M, K, lda, m0, k0, ar, ah, fullK, fa);
      if (!BT) load_b_regs(Bm, N, K, ldb, n0, k0, bkp, bn8, fullK, fb);
      else     load_bt_regs(Bm, N, K, ldb, n0, k0, btn, bth, fullK, fb);
    }

    // fragments: contiguous 32B per lane -> 2x ds_load_b128 each
    const v16bf a0 = *(const v16bf*)&lsA[cur][(2 * wm + 0) * 512 + lane * 16];
    const v16bf a1 = *(const v16bf*)&lsA[cur][(2 * wm + 1) * 512 + lane * 16];
    v16bf bb[4];
    #pragma unroll
    for (int j = 0; j < 4; ++j)
      bb[j] = *(const v16bf*)&lsB[cur][(4 * wn + j) * 512 + lane * 16];

    #pragma unroll
    for (int j = 0; j < 4; ++j) {
      acc[0][j] = __builtin_amdgcn_wmma_f32_16x16x32_bf16(false, a0, false, bb[j], (short)0, acc[0][j], false, false);
      acc[1][j] = __builtin_amdgcn_wmma_f32_16x16x32_bf16(false, a1, false, bb[j], (short)0, acc[1][j], false, false);
    }

    // store prefetched registers into the other buffer
    if (more) {
      store_a_lds(lsA[nxt], asub, amr, ah, fa);
      if (!BT) store_b_lds(lsB[nxt], bkp, bn8, fb);
      else     store_bt_lds(lsB[nxt], btn, bth, fb);
    }
    __syncthreads();
  }

  // ---- store: acc VGPR r -> M = r + 8*kh; lanes 0-15 = N ----
  #pragma unroll
  for (int i = 0; i < 2; ++i) {
    #pragma unroll
    for (int r = 0; r < 8; ++r) {
      int gm = m0 + wm * 32 + i * 16 + r + kh * 8;
      if (gm < M) {
        float* cp = C + (long)gm * ldc + n0 + wn * 64;
        #pragma unroll
        for (int j = 0; j < 4; ++j) {
          int cn = j * 16 + mi;
          if (n0 + wn * 64 + cn < N) cp[cn] = acc[i][j][r];
        }
      }
    }
  }
}

// ---------------------------------------------------------------------------
// Embedding: h[b,l,:] = tok_emb[x_t[b,l],:] + pos_emb[l,:]
// ---------------------------------------------------------------------------
__global__ void embed_kernel(const int* __restrict__ xt, const float* __restrict__ tok,
                             const float* __restrict__ pos, float* __restrict__ h)
{
  int idx = blockIdx.x * 256 + threadIdx.x;
  if (idx >= BL_ * D_) return;
  int bl = idx / D_, d = idx % D_;
  int l = bl % L_;
  int t = xt[bl];
  h[idx] = tok[(long)t * D_ + d] + pos[l * D_ + d];
}

// ---------------------------------------------------------------------------
// Timestep embedding -> c (B, COND). One block per batch element.
// ---------------------------------------------------------------------------
__global__ __launch_bounds__(256) void timestep_kernel(
    const float* __restrict__ sigma,
    const float* __restrict__ w1, const float* __restrict__ b1,
    const float* __restrict__ w2, const float* __restrict__ b2,
    float* __restrict__ c)
{
  __shared__ float emb[FREQ_];
  __shared__ float hid[COND_];
  int b = blockIdx.x, tid = threadIdx.x;
  if (tid < FREQ_ / 2) {
    float f = __expf(-logf(10000.0f) * (float)tid / (float)(FREQ_ / 2));
    float a = sigma[b] * f;
    emb[tid] = __cosf(a);
    emb[tid + FREQ_ / 2] = __sinf(a);
  }
  __syncthreads();
  if (tid < COND_) {
    float s = b1[tid];
    for (int k = 0; k < FREQ_; ++k) s += emb[k] * w1[k * COND_ + tid];
    hid[tid] = s / (1.0f + __expf(-s));   // silu
  }
  __syncthreads();
  if (tid < COND_) {
    float s = b2[tid];
    for (int k = 0; k < COND_; ++k) s += hid[k] * w2[k * COND_ + tid];
    c[b * COND_ + tid] = s;
  }
}

// ---------------------------------------------------------------------------
// mod = c @ w + bias   (B x 3D), one thread per output element (K=128 dot)
// ---------------------------------------------------------------------------
__global__ void mod_kernel(const float* __restrict__ c, const float* __restrict__ w,
                           const float* __restrict__ bias, float* __restrict__ mod)
{
  int idx = blockIdx.x * 256 + threadIdx.x;
  if (idx >= B_ * 3 * D_) return;
  int b = idx / (3 * D_), j = idx % (3 * D_);
  const float* cb = c + b * COND_;
  float s = bias[j];
  for (int k = 0; k < COND_; ++k) s += cb[k] * w[k * 3 * D_ + j];
  mod[idx] = s;
}

// ---------------------------------------------------------------------------
// AdaLN: out = LN(h) * (1+scale) + shift. One block per (b,l) row.
// ---------------------------------------------------------------------------
__global__ __launch_bounds__(256) void ln_mod_kernel(
    const float* __restrict__ h, const float* __restrict__ mod, float* __restrict__ out)
{
  __shared__ float red[256];
  int row = blockIdx.x;     // 0..BL-1
  int b = row / L_;
  int tid = threadIdx.x;
  const float* x = h + (long)row * D_;

  float s = 0.f;
  for (int d = tid; d < D_; d += 256) s += x[d];
  red[tid] = s; __syncthreads();
  for (int o = 128; o > 0; o >>= 1) { if (tid < o) red[tid] += red[tid + o]; __syncthreads(); }
  float mu = red[0] / (float)D_;
  __syncthreads();

  s = 0.f;
  for (int d = tid; d < D_; d += 256) { float t = x[d] - mu; s += t * t; }
  red[tid] = s; __syncthreads();
  for (int o = 128; o > 0; o >>= 1) { if (tid < o) red[tid] += red[tid + o]; __syncthreads(); }
  float rstd = rsqrtf(red[0] / (float)D_ + 1e-5f);

  const float* shift = mod + b * 3 * D_;
  const float* scale = shift + D_;
  float* op = out + (long)row * D_;
  for (int d = tid; d < D_; d += 256)
    op[d] = (x[d] - mu) * rstd * (1.0f + scale[d]) + shift[d];
}

// ---------------------------------------------------------------------------
// Flip along L
// ---------------------------------------------------------------------------
__global__ void flip_kernel(const float* __restrict__ in, float* __restrict__ out)
{
  int idx = blockIdx.x * 256 + threadIdx.x;
  if (idx >= BL_ * D_) return;
  int bl = idx / D_, d = idx % D_;
  int b = bl / L_, l = bl % L_;
  out[idx] = in[((long)(b * L_ + (L_ - 1 - l))) * D_ + d];
}

// ---------------------------------------------------------------------------
// delta = softplus(delta_pre + dt_bias[d])
// ---------------------------------------------------------------------------
__global__ void softplus_kernel(float* __restrict__ delta, const float* __restrict__ dtb)
{
  int idx = blockIdx.x * 256 + threadIdx.x;
  if (idx >= BL_ * DI_) return;
  int d = idx % DI_;
  float x = delta[idx] + dtb[d];
  delta[idx] = (x > 20.0f) ? x : log1pf(__expf(x));
}

// ---------------------------------------------------------------------------
// Selective-scan recurrence: one thread per (b, d) channel, N=16 state in regs.
// B/C vectors staged in LDS (8 timesteps at a time, shared across channels).
// ---------------------------------------------------------------------------
__global__ __launch_bounds__(256) void scan_kernel(
    const float* __restrict__ delta, const float* __restrict__ xz,
    const float* __restrict__ dtbc,  const float* __restrict__ A_log,
    const float* __restrict__ Dsk,   float* __restrict__ ys)
{
  const int CH = DI_ / 256;   // d-chunks per batch
  int b  = blockIdx.x / CH;
  int dc = blockIdx.x % CH;
  int d  = dc * 256 + threadIdx.x;

  float Aa[N_], hs[N_];
  float Dv = Dsk[d];
  #pragma unroll
  for (int n = 0; n < N_; ++n) { Aa[n] = -__expf(A_log[d * N_ + n]); hs[n] = 0.0f; }

  __shared__ float sBC[8][32];
  for (int t0 = 0; t0 < L_; t0 += 8) {
    int tt = threadIdx.x >> 5, j = threadIdx.x & 31;
    sBC[tt][j] = dtbc[((long)(b * L_ + t0 + tt)) * 80 + 48 + j];
    __syncthreads();
    #pragma unroll 2
    for (int q = 0; q < 8; ++q) {
      long rb = (long)(b * L_ + t0 + q);
      float dlt = delta[rb * DI_ + d];
      float u   = xz[rb * (2 * DI_) + d];
      float du  = dlt * u;
      float y   = 0.0f;
      #pragma unroll
      for (int n = 0; n < N_; ++n) {
        float dA = __expf(dlt * Aa[n]);
        hs[n] = dA * hs[n] + du * sBC[q][n];
        y += hs[n] * sBC[q][16 + n];
      }
      ys[rb * DI_ + d] = y + u * Dv;
    }
    __syncthreads();
  }
}

// ---------------------------------------------------------------------------
// ys *= silu(z), z = xz[..., DI+d]
// ---------------------------------------------------------------------------
__global__ void silu_gate_kernel(float* __restrict__ ys, const float* __restrict__ xz)
{
  int idx = blockIdx.x * 256 + threadIdx.x;
  if (idx >= BL_ * DI_) return;
  int bl = idx / DI_, d = idx % DI_;
  float z = xz[(long)bl * (2 * DI_) + DI_ + d];
  ys[idx] *= z / (1.0f + __expf(-z));
}

// ---------------------------------------------------------------------------
// h += gate * (yf + flip(yb))
// ---------------------------------------------------------------------------
__global__ void add_dual_kernel(float* __restrict__ h, const float* __restrict__ mod,
                                const float* __restrict__ yf, const float* __restrict__ yb)
{
  int idx = blockIdx.x * 256 + threadIdx.x;
  if (idx >= BL_ * D_) return;
  int bl = idx / D_, d = idx % D_;
  int b = bl / L_, l = bl % L_;
  float g  = mod[b * 3 * D_ + 2 * D_ + d];
  float vb = yb[((long)(b * L_ + (L_ - 1 - l))) * D_ + d];
  h[idx] += g * (yf[idx] + vb);
}

// t1 = silu(t1) * t2 over BL*H
__global__ void silu_mul_kernel(float* __restrict__ t1, const float* __restrict__ t2)
{
  int idx = blockIdx.x * 256 + threadIdx.x;
  if (idx >= BL_ * H_) return;
  float x = t1[idx];
  t1[idx] = (x / (1.0f + __expf(-x))) * t2[idx];
}

// h += gate * y
__global__ void add_single_kernel(float* __restrict__ h, const float* __restrict__ mod,
                                  const float* __restrict__ y)
{
  int idx = blockIdx.x * 256 + threadIdx.x;
  if (idx >= BL_ * D_) return;
  int bl = idx / D_, d = idx % D_;
  int b = bl / L_;
  h[idx] += mod[b * 3 * D_ + 2 * D_ + d] * y[idx];
}

// ---------------------------------------------------------------------------
// In-place log_softmax over V with MASK_ID forcing; one block per (b,l) row.
// ---------------------------------------------------------------------------
__global__ __launch_bounds__(256) void logsoftmax_kernel(float* __restrict__ out,
                                                         const int* __restrict__ xt)
{
  __shared__ float red[256];
  int row = blockIdx.x, tid = threadIdx.x;
  int tok = xt[row];
  float* p = out + (long)row * V_;

  if (tok != MASK_ID_) {   // unmasked: forced one-hot log-probs
    for (int v = tid; v < V_; v += 256) p[v] = (v == tok) ? 0.0f : NEGF;
    return;
  }
  float m = NEGF;
  for (int v = tid; v < V_; v += 256) {
    float x = (v == MASK_ID_) ? NEGF : p[v];
    m = fmaxf(m, x);
  }
  red[tid] = m; __syncthreads();
  for (int o = 128; o > 0; o >>= 1) { if (tid < o) red[tid] = fmaxf(red[tid], red[tid + o]); __syncthreads(); }
  m = red[0]; __syncthreads();

  float s = 0.0f;
  for (int v = tid; v < V_; v += 256) {
    float x = (v == MASK_ID_) ? NEGF : p[v];
    s += __expf(x - m);
  }
  red[tid] = s; __syncthreads();
  for (int o = 128; o > 0; o >>= 1) { if (tid < o) red[tid] += red[tid + o]; __syncthreads(); }
  float ls = logf(red[0]) + m;

  for (int v = tid; v < V_; v += 256) {
    float x = (v == MASK_ID_) ? NEGF : p[v];
    p[v] = x - ls;
  }
}

// ---------------------------------------------------------------------------
// Host launcher
// ---------------------------------------------------------------------------
extern "C" void kernel_launch(void* const* d_in, const int* in_sizes, int n_in,
                              void* d_out, int out_size, void* d_ws, size_t ws_size,
                              hipStream_t stream)
{
  (void)in_sizes; (void)n_in; (void)out_size; (void)ws_size;

  const int*   x_t       = (const int*)  d_in[0];
  const float* sigma     = (const float*)d_in[1];
  const float* tok_emb   = (const float*)d_in[2];
  const float* pos_emb   = (const float*)d_in[3];
  const float* te_w1     = (const float*)d_in[4];
  const float* te_b1     = (const float*)d_in[5];
  const float* te_w2     = (const float*)d_in[6];
  const float* te_b2     = (const float*)d_in[7];
  const float* adaln1_w  = (const float*)d_in[8];
  const float* adaln1_b  = (const float*)d_in[9];
  const float* adaln2_w  = (const float*)d_in[10];
  const float* adaln2_b  = (const float*)d_in[11];
  const float* Win       = (const float*)d_in[12];
  const float* Wx        = (const float*)d_in[13];
  const float* Wdt       = (const float*)d_in[14];
  const float* dt_bias   = (const float*)d_in[15];
  const float* A_log     = (const float*)d_in[16];
  const float* Dskip     = (const float*)d_in[17];
  const float* Wout      = (const float*)d_in[18];
  const float* mlp_w1    = (const float*)d_in[19];
  const float* mlp_w2    = (const float*)d_in[20];
  const float* mlp_w3    = (const float*)d_in[21];
  const float* outadaln_w = (const float*)d_in[22];
  const float* outadaln_b = (const float*)d_in[23];
  float* out = (float*)d_out;

  // workspace layout (floats)
  float* ws = (float*)d_ws;
  size_t off = 0;
  float* buf_h      = ws + off; off += (size_t)BL_ * D_;
  float* buf_c      = ws + off; off += (size_t)B_ * COND_;
  float* buf_mod    = ws + off; off += (size_t)B_ * 3 * D_;
  float* buf_normed = ws + off; off += (size_t)BL_ * D_;
  float* buf_flip   = ws + off; off += (size_t)BL_ * D_;
  float* buf_xz     = ws + off; off += (size_t)BL_ * 2 * DI_;
  float* buf_dtbc   = ws + off; off += (size_t)BL_ * 80;
  float* buf_delta  = ws + off; off += (size_t)BL_ * DI_;   // also MLP t1
  float* buf_ys     = ws + off; off += (size_t)BL_ * DI_;   // also MLP t2
  float* buf_yf     = ws + off; off += (size_t)BL_ * D_;    // also MLP out
  float* buf_yb     = ws + off; off += (size_t)BL_ * D_;

  auto gemm = [&](const float* A, const float* Bm, float* C,
                  int M, int N, int K, int lda, int ldb, int ldc, bool bt) {
    dim3 g((unsigned)((N + 127) / 128), (unsigned)((M + 127) / 128));
    if (bt) gemm_wmma_kernel<true ><<<g, 256, 0, stream>>>(A, Bm, C, M, N, K, lda, ldb, ldc);
    else    gemm_wmma_kernel<false><<<g, 256, 0, stream>>>(A, Bm, C, M, N, K, lda, ldb, ldc);
  };
  auto ew = [](int n) { return (unsigned)((n + 255) / 256); };

  embed_kernel<<<ew(BL_ * D_), 256, 0, stream>>>(x_t, tok_emb, pos_emb, buf_h);
  timestep_kernel<<<B_, 256, 0, stream>>>(sigma, te_w1, te_b1, te_w2, te_b2, buf_c);

  for (int l = 0; l < NL_; ++l) {
    // ---- SSM sublayer ----
    mod_kernel<<<ew(B_ * 3 * D_), 256, 0, stream>>>(
        buf_c, adaln1_w + (long)l * COND_ * 3 * D_, adaln1_b + (long)l * 3 * D_, buf_mod);
    ln_mod_kernel<<<BL_, 256, 0, stream>>>(buf_h, buf_mod, buf_normed);

    for (int dr = 0; dr < 2; ++dr) {
      const float* src = buf_normed;
      if (dr == 1) {
        flip_kernel<<<ew(BL_ * D_), 256, 0, stream>>>(buf_normed, buf_flip);
        src = buf_flip;
      }
      long wi = (long)l * 2 + dr;
      // xz = src @ Win  (768 -> 3072)
      gemm(src, Win + wi * D_ * 2 * DI_, buf_xz, BL_, 2 * DI_, D_, D_, 2 * DI_, 2 * DI_, false);
      // dtbc = u @ Wx   (u strided inside xz; 1536 -> 80)
      gemm(buf_xz, Wx + wi * DI_ * (DTR_ + 2 * N_), buf_dtbc, BL_, DTR_ + 2 * N_, DI_,
           2 * DI_, DTR_ + 2 * N_, DTR_ + 2 * N_, false);
      // delta_pre = dt_low @ Wdt (K=48, strided A)
      gemm(buf_dtbc, Wdt + wi * DTR_ * DI_, buf_delta, BL_, DI_, DTR_, 80, DI_, DI_, false);
      softplus_kernel<<<ew(BL_ * DI_), 256, 0, stream>>>(buf_delta, dt_bias + wi * DI_);
      scan_kernel<<<B_ * (DI_ / 256), 256, 0, stream>>>(
          buf_delta, buf_xz, buf_dtbc, A_log + wi * DI_ * N_, Dskip + wi * DI_, buf_ys);
      silu_gate_kernel<<<ew(BL_ * DI_), 256, 0, stream>>>(buf_ys, buf_xz);
      // y @ Wout (1536 -> 768)
      gemm(buf_ys, Wout + wi * DI_ * D_, (dr == 0 ? buf_yf : buf_yb),
           BL_, D_, DI_, DI_, D_, D_, false);
    }
    add_dual_kernel<<<ew(BL_ * D_), 256, 0, stream>>>(buf_h, buf_mod, buf_yf, buf_yb);

    // ---- MLP sublayer ----
    mod_kernel<<<ew(B_ * 3 * D_), 256, 0, stream>>>(
        buf_c, adaln2_w + (long)l * COND_ * 3 * D_, adaln2_b + (long)l * 3 * D_, buf_mod);
    ln_mod_kernel<<<BL_, 256, 0, stream>>>(buf_h, buf_mod, buf_normed);
    gemm(buf_normed, mlp_w1 + (long)l * D_ * H_, buf_delta, BL_, H_, D_, D_, H_, H_, false);
    gemm(buf_normed, mlp_w2 + (long)l * D_ * H_, buf_ys,    BL_, H_, D_, D_, H_, H_, false);
    silu_mul_kernel<<<ew(BL_ * H_), 256, 0, stream>>>(buf_delta, buf_ys);
    gemm(buf_delta, mlp_w3 + (long)l * H_ * D_, buf_yf, BL_, D_, H_, H_, D_, D_, false);
    add_single_kernel<<<ew(BL_ * D_), 256, 0, stream>>>(buf_h, buf_mod, buf_yf);
  }

  // ---- output head ----
  mod_kernel<<<ew(B_ * 3 * D_), 256, 0, stream>>>(buf_c, outadaln_w, outadaln_b, buf_mod);
  ln_mod_kernel<<<BL_, 256, 0, stream>>>(buf_h, buf_mod, buf_normed);
  // logits = normed @ tok_emb^T  (B-matrix transposed)
  gemm(buf_normed, tok_emb, out, BL_, V_, D_, D_, D_, V_, true);
  logsoftmax_kernel<<<BL_, 256, 0, stream>>>(out, x_t);
}
